// Block_21938692948274
// MI455X (gfx1250) — matrix operand order repaired and checked by reading, not compiled, not run on hardware
//
#include <hip/hip_runtime.h>

typedef __attribute__((ext_vector_type(16))) _Float16 v16h;
typedef __attribute__((ext_vector_type(8)))  float    v8f;

namespace k455 {

constexpr int  B_   = 2;
constexpr int  C_   = 64;
constexpr int  H_   = 192;
constexpr int  W_   = 192;
constexpr long HW_  = 36864;
constexpr int  NS_  = 144;
constexpr int  NWS_ = 12;
constexpr int  W2_  = 97;
constexpr long HW2_ = 18624;   // 192*97

// ---------------------------------------------------------------------------
// Generic wave-level WMMA GEMM:  C[m,n] (+)= alpha * sum_k A[m,k]*B[k,n]
// In-batch strides are 32-bit (all tensors < 2^31 elements); batch offsets 64-bit.
// mode: 0 = store, 1 = read-add-write, 2 = atomicAdd (split-K)
// Loads are branch-free: full tiles take an unconditional path, edge tiles /
// K-tails load from clamped addresses and select to zero (no exec branching,
// so VMEM clauses batch and EXEC stays all-ones for WMMA).
// ---------------------------------------------------------------------------
__global__ void gemm_wmma(const float* __restrict__ A, const float* __restrict__ Bm,
                          float* __restrict__ C,
                          int M, int N, int K,
                          int sam, int sak, int sbk, int sbn, int scm, int scn,
                          long bA, long bB, long bC,
                          float alpha, int mode, int kchunk)
{
  const int lane = threadIdx.x & 31;
  const int half = lane >> 4;
  const int mn   = lane & 15;
  const int ntN  = (N + 15) >> 4;
  const int m0   = (int)(blockIdx.x / ntN) << 4;
  const int n0   = (int)(blockIdx.x % ntN) << 4;
  const float* Ab = A  + (long)blockIdx.z * bA;
  const float* Bb = Bm + (long)blockIdx.z * bB;
  float*       Cb = C  + (long)blockIdx.z * bC;
  const int kbeg = blockIdx.y * kchunk;
  const int kend = (kbeg + kchunk < K) ? (kbeg + kchunk) : K;
  const int m = m0 + mn;
  const int n = n0 + mn;
  const bool mok = m < M;
  const bool nok = n < N;
  const bool full = (m0 + 16 <= M) && (n0 + 16 <= N);

  // Row/column base pointers (clamped so every load address is valid).
  const float* pa = Ab + (long)(mok ? m : 0) * sam;
  const float* pb = Bb + (long)(nok ? n : 0) * sbn;

  // Per-fragment K offsets (A: 16-bit 16x32 layout, B: 16-bit 32x16 layout).
  int kaoff[8], kboff[8];
#pragma unroll
  for (int j = 0; j < 8; ++j) {
    kaoff[j] = ((j >> 2) << 4) + (half << 3) + ((j & 3) << 1);
    kboff[j] = ((j >> 2) << 4) + (half << 3) + ((j & 3) << 1);
  }

  v8f acc = {};
  for (int k0 = kbeg; k0 < kend; k0 += 32) {
    float fa[16], fb[16];
    if (full && (k0 + 32 <= kend)) {
      // ---- fast path: unconditional loads ----
#pragma unroll
      for (int j = 0; j < 8; ++j) {
        int ka = k0 + kaoff[j];
        int kb = k0 + kboff[j];
        fa[2*j]   = pa[ka * sak];
        fa[2*j+1] = pa[(ka + 1) * sak];
        fb[2*j]   = pb[kb * sbk];
        fb[2*j+1] = pb[(kb + 1) * sbk];
      }
    } else {
      // ---- edge path: clamped addresses + value select (no branches) ----
#pragma unroll
      for (int j = 0; j < 8; ++j) {
        int ka = k0 + kaoff[j];
        int kb = k0 + kboff[j];
        int ka0 = ka     < kend ? ka     : kend - 1;
        int ka1 = ka + 1 < kend ? ka + 1 : kend - 1;
        int kb0 = kb     < kend ? kb     : kend - 1;
        int kb1 = kb + 1 < kend ? kb + 1 : kend - 1;
        float a0 = pa[ka0 * sak];
        float a1 = pa[ka1 * sak];
        float b0 = pb[kb0 * sbk];
        float b1 = pb[kb1 * sbk];
        fa[2*j]   = (mok && ka     < kend) ? a0 : 0.f;
        fa[2*j+1] = (mok && ka + 1 < kend) ? a1 : 0.f;
        fb[2*j]   = (nok && kb     < kend) ? b0 : 0.f;
        fb[2*j+1] = (nok && kb + 1 < kend) ? b1 : 0.f;
      }
    }
    v16h af, bf;
#pragma unroll
    for (int e = 0; e < 16; ++e) { af[e] = (_Float16)fa[e]; bf[e] = (_Float16)fb[e]; }
    acc = __builtin_amdgcn_wmma_f32_16x16x32_f16(false, af, false, bf, (short)0, acc,
                                                 false, false);
  }
#pragma unroll
  for (int r = 0; r < 8; ++r) {
    int mm = m0 + (half << 3) + r;
    if (mm < M && nok) {
      long idx = (long)mm * scm + (long)n * scn;
      float v = alpha * acc[r];
      if (mode == 0)      Cb[idx]  = v;
      else if (mode == 1) Cb[idx] += v;
      else                atomicAdd(&Cb[idx], v);
    }
  }
}

// ---------------------------------------------------------------------------
// SSN distance + softmax + scatter. One block per (batch, superpixel tile).
// 16 waves; wave w computes dot(9 candidates x 16 pixels of row w) via WMMA.
// ---------------------------------------------------------------------------
__global__ void ssn_dist(const float* __restrict__ comb, const float* __restrict__ cent,
                         float* __restrict__ aff)
{
  const int blk = blockIdx.x;
  const int b  = blk / NS_;
  const int t  = blk % NS_;
  const int tr = t / NWS_, tc = t % NWS_;
  __shared__ float s_dot[16][16][17];
  __shared__ float s_s2[16];
  const int tid  = threadIdx.x;
  const int wave = tid >> 5;
  const int lane = tid & 31;
  const int half = lane >> 4;
  const int mn   = lane & 15;
  const float* combB = comb + (long)b * 128 * HW_;
  const float* centB = cent + (long)b * 128 * NS_;

  if (tid < 16) {
    float s2 = 0.f;
    if (tid < 9) {
      int cs = t + (tid / 3 - 1) * NWS_ + (tid % 3 - 1);
      cs = cs < 0 ? 0 : (cs > NS_ - 1 ? NS_ - 1 : cs);
      for (int k = 0; k < 128; ++k) {
        float wv = (k < 64) ? 1.f : 10.f;
        float cv = centB[k * NS_ + cs];
        s2 += wv * cv * cv;
      }
    }
    s_s2[tid] = s2;
  }

  const bool mval = mn < 9;
  int cs_m = t + ((mn < 9 ? mn : 0) / 3 - 1) * NWS_ + ((mn < 9 ? mn : 0) % 3 - 1);
  cs_m = cs_m < 0 ? 0 : (cs_m > NS_ - 1 ? NS_ - 1 : cs_m);
  const int p = (tr * 16 + wave) * W_ + tc * 16 + mn;
  const float* pcent = centB + cs_m;   // always-valid base
  const float* pcomb = combB + p;

  v8f acc = {};
  for (int k0 = 0; k0 < 128; k0 += 32) {
    float fa[16], fb[16];
#pragma unroll
    for (int j = 0; j < 8; ++j) {
      int ka = k0 + ((j >> 2) << 4) + (half << 3) + ((j & 3) << 1);
      int kb = k0 + ((j >> 2) << 4) + (half << 3) + ((j & 3) << 1);
      float a0 = pcent[ka * NS_];
      float a1 = pcent[(ka + 1) * NS_];
      fa[2*j]   = mval ? ((ka     < 64) ? a0 : 10.f * a0) : 0.f;
      fa[2*j+1] = mval ? ((ka + 1 < 64) ? a1 : 10.f * a1) : 0.f;
      fb[2*j]   = pcomb[kb * (int)HW_];
      fb[2*j+1] = pcomb[(kb + 1) * (int)HW_];
    }
    v16h af, bf;
#pragma unroll
    for (int e = 0; e < 16; ++e) { af[e] = (_Float16)fa[e]; bf[e] = (_Float16)fb[e]; }
    acc = __builtin_amdgcn_wmma_f32_16x16x32_f16(false, af, false, bf, (short)0, acc,
                                                 false, false);
  }
#pragma unroll
  for (int r = 0; r < 8; ++r) s_dot[wave][(half << 3) + r][mn] = acc[r];
  __syncthreads();

  if (tid < 256) {
    const int i = tid >> 4, j = tid & 15;
    const int pp = (tr * 16 + i) * W_ + tc * 16 + j;
    float p2 = 0.f;
    for (int k = 0; k < 128; ++k) {
      float v = combB[k * (int)HW_ + pp];
      p2 += ((k < 64) ? 1.f : 10.f) * v * v;
    }
    float nd[9]; bool val[9]; float mx = -3.0e38f;
#pragma unroll
    for (int jj = 0; jj < 9; ++jj) {
      int dy = jj / 3 - 1, dx = jj % 3 - 1;
      val[jj] = (tr + dy >= 0) && (tr + dy < NWS_) && (tc + dx >= 0) && (tc + dx < NWS_);
      float d = p2 + s_s2[jj] - 2.f * s_dot[i][jj][j];
      nd[jj] = val[jj] ? -d : -3.0e38f;
      mx = fmaxf(mx, nd[jj]);
    }
    float z = 0.f, e[9];
#pragma unroll
    for (int jj = 0; jj < 9; ++jj) { e[jj] = val[jj] ? expf(nd[jj] - mx) : 0.f; z += e[jj]; }
    float iz = 1.f / z;
#pragma unroll
    for (int jj = 0; jj < 9; ++jj) {
      if (val[jj]) {
        int cs = t + (jj / 3 - 1) * NWS_ + (jj % 3 - 1);
        aff[((long)b * NS_ + cs) * HW_ + pp] = e[jj] * iz;
      }
    }
  }
}

// ---------------------------------------------------------------------------
// Small utility kernels
// ---------------------------------------------------------------------------
__global__ void zero_kernel(float* __restrict__ p, long n)
{
  long i = (long)blockIdx.x * blockDim.x + threadIdx.x;
  if (i < n) p[i] = 0.f;
}

__global__ void fold_w_kernel(const float* __restrict__ cdw, float* __restrict__ w2)
{
  int i = blockIdx.x * blockDim.x + threadIdx.x;
  if (i >= C_ * C_) return;
  const float* src = cdw + (long)i * 9;
  float s = 0.f;
  for (int t = 0; t < 9; ++t) s += src[t];
  for (int t = 0; t < 9; ++t) w2[(long)i * 9 + t] = src[t] - ((t == 4) ? 0.7f * s : 0.f);
}

__global__ void copy_x_to_comb(const float* __restrict__ x, float* __restrict__ comb)
{
  long i = (long)blockIdx.x * blockDim.x + threadIdx.x;
  if (i >= (long)B_ * C_ * HW_) return;
  long b = i / (C_ * HW_), r = i % (C_ * HW_);
  comb[b * 128 * HW_ + r] = x[i];
}

__global__ void im2col_kernel(const float* __restrict__ x, float* __restrict__ col)
{
  long i = (long)blockIdx.x * blockDim.x + threadIdx.x;
  if (i >= (long)B_ * 576 * HW_) return;
  long b = i / (576 * HW_), r = i % (576 * HW_);
  int  k = (int)(r / HW_);
  int  p = (int)(r % HW_);
  int  c = k / 9, t = k % 9;
  int  h = p / W_, w = p % W_;
  int  h2 = h + t / 3 - 1, w2 = w + t % 3 - 1;
  float v = 0.f;
  if (h2 >= 0 && h2 < H_ && w2 >= 0 && w2 < W_)
    v = x[((long)b * C_ + c) * HW_ + h2 * W_ + w2];
  col[i] = v;
}

__global__ void cent_init_kernel(const float* __restrict__ comb, float* __restrict__ cent)
{
  int i = blockIdx.x * blockDim.x + threadIdx.x;
  if (i >= B_ * 128 * NS_) return;
  int b = i / (128 * NS_), r = i % (128 * NS_);
  int k = r / NS_, s = r % NS_;
  int tr = s / NWS_, tc = s % NWS_;
  const float* base = comb + ((long)b * 128 + k) * HW_ + (tr * 16) * W_ + tc * 16;
  float sum = 0.f;
  for (int ii = 0; ii < 16; ++ii)
    for (int jj = 0; jj < 16; ++jj) sum += base[ii * W_ + jj];
  cent[i] = sum * (1.f / 256.f);
}

__global__ void rowsum_kernel(const float* __restrict__ data, float* __restrict__ out, long rowlen)
{
  __shared__ float red[256];
  const float* r = data + (long)blockIdx.x * rowlen;
  const int tid = threadIdx.x;
  float s = 0.f;
  for (long i = tid; i < rowlen; i += 256) s += r[i];
  red[tid] = s; __syncthreads();
  for (int o = 128; o > 0; o >>= 1) { if (tid < o) red[tid] += red[tid + o]; __syncthreads(); }
  if (tid == 0) out[blockIdx.x] = red[0];
}

__global__ void cent_update_kernel(const float* __restrict__ cn, const float* __restrict__ rs,
                                   float* __restrict__ cent)
{
  int i = blockIdx.x * blockDim.x + threadIdx.x;
  if (i >= B_ * 128 * NS_) return;
  int b = i / (128 * NS_), r = i % (128 * NS_);
  int k = r / NS_, s = r % NS_;
  cent[i] = cn[((long)b * NS_ + s) * 128 + k] / (rs[b * NS_ + s] + 1e-16f);
}

__global__ void stoken_div_kernel(float* __restrict__ st, const float* __restrict__ rs)
{
  int i = blockIdx.x * blockDim.x + threadIdx.x;
  if (i >= B_ * NS_ * C_) return;
  st[i] /= (rs[i / C_] + 1e-16f);
}

__global__ void layernorm_kernel(const float* __restrict__ x, const float* __restrict__ g,
                                 const float* __restrict__ bb, float* __restrict__ xn)
{
  long i = (long)blockIdx.x * blockDim.x + threadIdx.x;
  if (i >= (long)B_ * HW_) return;
  long b = i / HW_;
  int  p = (int)(i % HW_);
  const float* xb = x + b * C_ * HW_ + p;
  float mu = 0.f;
  for (int c = 0; c < C_; ++c) mu += xb[c * (int)HW_];
  mu *= (1.f / C_);
  float var = 0.f;
  for (int c = 0; c < C_; ++c) { float d = xb[c * (int)HW_] - mu; var += d * d; }
  var *= (1.f / C_);
  float inv = rsqrtf(var + 1e-6f);
  float* xo = xn + b * C_ * HW_ + p;
  for (int c = 0; c < C_; ++c)
    xo[c * (int)HW_] = (xb[c * (int)HW_] - mu) * inv * g[c] + bb[c];
}

__global__ void softmax_rows(float* __restrict__ data, long rowlen)
{
  __shared__ float red[256];
  float* r = data + (long)blockIdx.x * rowlen;
  const int tid = threadIdx.x;
  float m = -3.0e38f;
  for (long i = tid; i < rowlen; i += 256) m = fmaxf(m, r[i]);
  red[tid] = m; __syncthreads();
  for (int o = 128; o > 0; o >>= 1) { if (tid < o) red[tid] = fmaxf(red[tid], red[tid + o]); __syncthreads(); }
  const float mx = red[0]; __syncthreads();
  float s = 0.f;
  for (long i = tid; i < rowlen; i += 256) s += expf(r[i] - mx);
  red[tid] = s; __syncthreads();
  for (int o = 128; o > 0; o >>= 1) { if (tid < o) red[tid] += red[tid + o]; __syncthreads(); }
  const float inv = 1.f / red[0];
  for (long i = tid; i < rowlen; i += 256) r[i] = expf(r[i] - mx) * inv;
}

__global__ void dwconv3x3_kernel(const float* __restrict__ in, const float* __restrict__ w,
                                 const float* __restrict__ bias, float* __restrict__ out,
                                 int Cc, int Hh, int Ww, int res)
{
  long i = (long)blockIdx.x * blockDim.x + threadIdx.x;
  long tot = (long)B_ * Cc * Hh * Ww;
  if (i >= tot) return;
  int  wq = (int)(i % Ww); long t = i / Ww;
  int  hq = (int)(t % Hh); t /= Hh;
  int  c  = (int)(t % Cc);
  long bc = i / ((long)Hh * Ww);
  const float* ib = in + bc * Hh * Ww;
  float s = bias[c];
  for (int dh = -1; dh <= 1; ++dh)
    for (int dw = -1; dw <= 1; ++dw) {
      int h2 = hq + dh, w2 = wq + dw;
      if (h2 >= 0 && h2 < Hh && w2 >= 0 && w2 < Ww)
        s += w[c * 9 + (dh + 1) * 3 + (dw + 1)] * ib[h2 * Ww + w2];
    }
  if (res) s += ib[hq * Ww + wq];
  out[i] = s;
}

__global__ void add3_kernel(const float* __restrict__ x, const float* __restrict__ xout,
                            const float* __restrict__ vconv, float* __restrict__ x2)
{
  long i = (long)blockIdx.x * blockDim.x + threadIdx.x;
  if (i >= (long)B_ * C_ * HW_) return;
  long b = i / (C_ * HW_);
  int  r = (int)(i % (C_ * HW_));
  float lepe = vconv[b * C_ * HW_ + (r % C_) * (int)HW_ + (r / C_)];
  x2[i] = x[i] + xout[i] + lepe;
}

__global__ void basis_init_kernel(float* Cw, float* Sw, float* Ch, float* Sh,
                                  float* Cw2, float* Sw2)
{
  int i = blockIdx.x * blockDim.x + threadIdx.x;
  const float tw = 6.28318530717958647692f / 192.f;
  if (i < 192 * 97) {
    int w = i / 97, k = i % 97; int m = (w * k) % 192; float a = tw * m;
    Cw[i] = cosf(a); Sw[i] = sinf(a);
  }
  if (i < 192 * 192) {
    int h = i / 192, j = i % 192; int m = (h * j) % 192; float a = tw * m;
    Ch[i] = cosf(a); Sh[i] = sinf(a);
  }
  if (i < 97 * 192) {
    int k = i / 192, w = i % 192; int m = (k * w) % 192; float a = tw * m;
    float wk = (k == 0 || k == 96) ? 1.f : 2.f;
    Cw2[i] = wk * cosf(a); Sw2[i] = wk * sinf(a);
  }
}

__global__ void fftpack_kernel(const float* __restrict__ Fr, const float* __restrict__ Fi,
                               const float* __restrict__ bg, const float* __restrict__ bb,
                               float* __restrict__ frb)
{
  long i = (long)blockIdx.x * blockDim.x + threadIdx.x;
  if (i >= (long)B_ * 128 * HW2_) return;
  long b = i / (128 * HW2_), r = i % (128 * HW2_);
  int  ch = (int)(r / HW2_);
  int  p = (int)(r % HW2_);
  const float* src = (ch & 1) ? Fi : Fr;
  float v = src[(b * 64 + (ch >> 1)) * HW2_ + p];
  float inv = rsqrtf(1.00001f);
  frb[i] = v * inv * bg[ch] + bb[ch];
}

__global__ void gelu_deint_kernel(const float* __restrict__ fd, const float* __restrict__ bias,
                                  float* __restrict__ Gr, float* __restrict__ Gi)
{
  long i = (long)blockIdx.x * blockDim.x + threadIdx.x;
  if (i >= (long)B_ * 128 * HW2_) return;
  long b = i / (128 * HW2_), r = i % (128 * HW2_);
  int  ch = (int)(r / HW2_);
  int  p = (int)(r % HW2_);
  float v = fd[i] + bias[ch];
  float g = 0.5f * v * (1.f + erff(v * 0.70710678118654752f));
  float* dst = (ch & 1) ? Gi : Gr;
  dst[(b * 64 + (ch >> 1)) * HW2_ + p] = g;
}

__global__ void final_kernel(const float* __restrict__ xctmp, const float* __restrict__ xcb,
                             const float* __restrict__ g2, const float* __restrict__ b2,
                             const float* __restrict__ outf, float* __restrict__ out)
{
  long i = (long)blockIdx.x * blockDim.x + threadIdx.x;
  if (i >= (long)B_ * C_ * HW_) return;
  int c = (int)((i / HW_) % C_);
  float inv = rsqrtf(1.00001f);
  float v = (xctmp[i] + xcb[c]) * inv * g2[c] + b2[c];
  out[i] = fmaxf(v, 0.f) + outf[i];
}

} // namespace k455

// ---------------------------------------------------------------------------
// Host side
// ---------------------------------------------------------------------------
static void launch_gemm(hipStream_t s, const float* A, const float* B, float* C,
                        int M, int N, int K,
                        int sam, int sak, int sbk, int sbn, int scm, int scn,
                        long bA, long bB, long bC, int batch,
                        float alpha, int mode, int kchunk)
{
  int mt = (M + 15) / 16, nt = (N + 15) / 16;
  int ky = (K + kchunk - 1) / kchunk;
  dim3 grid((unsigned)(mt * nt), (unsigned)ky, (unsigned)batch);
  k455::gemm_wmma<<<grid, 32, 0, s>>>(A, B, C, M, N, K, sam, sak, sbk, sbn, scm, scn,
                                      bA, bB, bC, alpha, (ky > 1) ? 2 : mode, kchunk);
}

static inline unsigned nblk(long n) { return (unsigned)((n + 255) / 256); }

extern "C" void kernel_launch(void* const* d_in, const int* in_sizes, int n_in,
                              void* d_out, int out_size, void* d_ws, size_t ws_size,
                              hipStream_t stream)
{
  using namespace k455;
  (void)in_sizes; (void)n_in; (void)out_size; (void)ws_size;

  const float* x      = (const float*)d_in[0];
  const float* cd_w   = (const float*)d_in[1];
  const float* ln_g   = (const float*)d_in[2];
  const float* ln_b   = (const float*)d_in[3];
  const float* q_w    = (const float*)d_in[4];
  const float* k_w    = (const float*)d_in[5];
  const float* v_w    = (const float*)d_in[6];
  const float* sp_w   = (const float*)d_in[7];
  const float* lepe_w = (const float*)d_in[8];
  const float* lepe_b = (const float*)d_in[9];
  const float* bn_g   = (const float*)d_in[10];
  const float* bn_b   = (const float*)d_in[11];
  const float* fpe_w  = (const float*)d_in[12];
  const float* fpe_b  = (const float*)d_in[13];
  // d_in[14] wt_w, d_in[15] wt_b: softmax over a single channel == 1.0 -> unused
  const float* fdc_w  = (const float*)d_in[16];
  const float* fdc_b  = (const float*)d_in[17];
  const float* xc_w   = (const float*)d_in[18];
  const float* xc_b   = (const float*)d_in[19];
  const float* bn2_g  = (const float*)d_in[20];
  const float* bn2_b  = (const float*)d_in[21];
  float* out = (float*)d_out;

  const long E_img  = (long)B_ * C_ * HW_;        // 4,718,592
  const long E_comb = (long)B_ * 128 * HW_;       // 9,437,184
  const long E_aff  = (long)B_ * NS_ * HW_;       // 10,616,832
  const long E_big  = (long)B_ * 576 * HW_;       // 42,467,328 (= B*NH*NS*HW)
  const long E_fh   = (long)B_ * C_ * HW2_;       // 2,383,872
  const long E_f2   = (long)B_ * 128 * HW2_;      // 4,767,744

  float* ws = (float*)d_ws;
  long off = 0;
  auto alloc = [&](long n) { float* p = ws + off; off += (n + 63) & ~63L; return p; };

  float* x2     = alloc(E_img);
  float* xn     = alloc(E_img);         // later reused as x_out
  float* q      = alloc(E_img);         // later reused as xc tmp
  float* kk     = alloc(E_img);
  float* v      = alloc(E_img);
  float* vconv  = alloc(E_img);
  float* comb   = alloc(E_comb);
  float* aff    = alloc(E_aff);
  float* big    = alloc(E_big);         // im2col -> attn logits -> FFT temps
  float* w2     = alloc((long)C_ * C_ * 9);
  float* cent   = alloc((long)B_ * 128 * NS_);
  float* centn  = alloc((long)B_ * NS_ * 128);
  float* rsum   = alloc((long)B_ * NS_);
  float* stoken = alloc((long)B_ * NS_ * C_);
  float* stok   = alloc((long)B_ * C_ * NS_);
  float* s_out  = alloc((long)B_ * 4 * 16 * NS_);
  float* Cw  = alloc(192 * 97);
  float* Sw  = alloc(192 * 97);
  float* Ch  = alloc(192 * 192);
  float* Sh  = alloc(192 * 192);
  float* Cw2 = alloc(97 * 192);
  float* Sw2 = alloc(97 * 192);

  // FFT temporaries carved out of `big` (attention logits are dead by then)
  float* R1   = big;
  float* I1   = R1 + E_fh;
  float* Fr   = I1 + E_fh;
  float* Fi   = Fr + E_fh;
  float* frb  = Fi + E_fh;
  float* frb2 = frb + E_f2;
  float* fdt  = frb2 + E_f2;
  float* Gr   = fdt + E_f2;
  float* Gi   = Gr + E_fh;
  float* R1i  = Gi + E_fh;
  float* I1i  = R1i + E_fh;
  float* outf = I1i + E_fh;

  // ---- central-difference conv (folded into one 3x3 conv) -> comb[64:128] ----
  basis_init_kernel<<<nblk(192 * 192), 256, 0, stream>>>(Cw, Sw, Ch, Sh, Cw2, Sw2);
  fold_w_kernel<<<nblk(C_ * C_), 256, 0, stream>>>(cd_w, w2);
  copy_x_to_comb<<<nblk(E_img), 256, 0, stream>>>(x, comb);
  im2col_kernel<<<nblk(E_big), 256, 0, stream>>>(x, big);
  launch_gemm(stream, w2, big, comb + 64 * HW_, C_, (int)HW_, 576,
              576, 1, (int)HW_, 1, (int)HW_, 1, 0, 576 * HW_, 128 * HW_, B_, 1.f, 0, 576);

  // ---- SSN superpixel iterations ----
  cent_init_kernel<<<nblk(B_ * 128 * NS_), 256, 0, stream>>>(comb, cent);
  for (int it = 0; it < 2; ++it) {
    zero_kernel<<<nblk(E_aff), 256, 0, stream>>>(aff, E_aff);
    ssn_dist<<<dim3(B_ * NS_), 512, 0, stream>>>(comb, cent, aff);
    if (it == 0) {
      zero_kernel<<<nblk((long)B_ * NS_ * 128), 256, 0, stream>>>(centn, (long)B_ * NS_ * 128);
      launch_gemm(stream, aff, comb, centn, NS_, 128, (int)HW_,
                  (int)HW_, 1, 1, (int)HW_, 128, 1, (long)NS_ * HW_, 128 * HW_, (long)NS_ * 128,
                  B_, 1.f, 2, 2304);
      rowsum_kernel<<<B_ * NS_, 256, 0, stream>>>(aff, rsum, HW_);
      cent_update_kernel<<<nblk(B_ * 128 * NS_), 256, 0, stream>>>(centn, rsum, cent);
    }
  }

  // ---- SGCA ----
  layernorm_kernel<<<nblk((long)B_ * HW_), 256, 0, stream>>>(x, ln_g, ln_b, xn);
  launch_gemm(stream, q_w, xn, q, C_, (int)HW_, C_,
              C_, 1, (int)HW_, 1, (int)HW_, 1, 0, C_ * HW_, C_ * HW_, B_, 1.f, 0, C_);
  launch_gemm(stream, k_w, xn, kk, C_, (int)HW_, C_,
              C_, 1, (int)HW_, 1, (int)HW_, 1, 0, C_ * HW_, C_ * HW_, B_, 1.f, 0, C_);
  launch_gemm(stream, v_w, xn, v, C_, (int)HW_, C_,
              C_, 1, (int)HW_, 1, (int)HW_, 1, 0, C_ * HW_, C_ * HW_, B_, 1.f, 0, C_);
  dwconv3x3_kernel<<<nblk(E_img), 256, 0, stream>>>(v, lepe_w, lepe_b, vconv, C_, H_, W_, 0);

  rowsum_kernel<<<B_ * NS_, 256, 0, stream>>>(aff, rsum, HW_);
  zero_kernel<<<nblk((long)B_ * NS_ * C_), 256, 0, stream>>>(stoken, (long)B_ * NS_ * C_);
  launch_gemm(stream, aff, xn, stoken, NS_, C_, (int)HW_,
              (int)HW_, 1, 1, (int)HW_, C_, 1, (long)NS_ * HW_, C_ * HW_, (long)NS_ * C_,
              B_, 1.f, 2, 2304);
  stoken_div_kernel<<<nblk((long)B_ * NS_ * C_), 256, 0, stream>>>(stoken, rsum);
  launch_gemm(stream, sp_w, stoken, stok, C_, NS_, C_,
              C_, 1, 1, C_, NS_, 1, 0, (long)NS_ * C_, (long)C_ * NS_, B_, 1.f, 0, C_);

  const float scale = 0.25f;  // (C/NH)^-0.5
  // s_attn logits, stored transposed (s, p) for contiguous softmax over p
  launch_gemm(stream, stok, kk, big, NS_, (int)HW_, 16,
              1, NS_, (int)HW_, 1, (int)HW_, 1, 16L * NS_, 16L * HW_, (long)NS_ * HW_,
              B_ * 4, scale, 0, 32);
  softmax_rows<<<B_ * 4 * NS_, 256, 0, stream>>>(big, HW_);
  zero_kernel<<<nblk((long)B_ * 4 * 16 * NS_), 256, 0, stream>>>(s_out, (long)B_ * 4 * 16 * NS_);
  launch_gemm(stream, v, big, s_out, 16, NS_, (int)HW_,
              (int)HW_, 1, 1, (int)HW_, NS_, 1, 16L * HW_, (long)NS_ * HW_, 16L * NS_,
              B_ * 4, 1.f, 2, 2304);
  // x_attn logits (p, s) for contiguous softmax over s
  launch_gemm(stream, q, stok, big, (int)HW_, NS_, 16,
              1, (int)HW_, NS_, 1, NS_, 1, 16L * HW_, 16L * NS_, (long)HW_ * NS_,
              B_ * 4, scale, 0, 32);
  softmax_rows<<<(unsigned)(B_ * 4 * HW_), 256, 0, stream>>>(big, NS_);
  // x_out into xn buffer (xn no longer needed)
  launch_gemm(stream, s_out, big, xn, 16, (int)HW_, NS_,
              NS_, 1, 1, NS_, (int)HW_, 1, 16L * NS_, (long)HW_ * NS_, 16L * HW_,
              B_ * 4, 1.f, 0, NS_);
  add3_kernel<<<nblk(E_img), 256, 0, stream>>>(x, xn, vconv, x2);

  // ---- DFFFN: rfft2 as DFT GEMMs ----
  launch_gemm(stream, x2, Cw, R1, B_ * C_ * H_, W2_, W_,
              W_, 1, W2_, 1, W2_, 1, 0, 0, 0, 1, 1.f, 0, W_);
  launch_gemm(stream, x2, Sw, I1, B_ * C_ * H_, W2_, W_,
              W_, 1, W2_, 1, W2_, 1, 0, 0, 0, 1, -1.f, 0, W_);
  const float fs = 1.f / 192.f;  // ortho norm
  launch_gemm(stream, Ch, R1, Fr, H_, W2_, H_, 192, 1, W2_, 1, W2_, 1,
              0, HW2_, HW2_, B_ * C_, fs, 0, H_);
  launch_gemm(stream, Sh, I1, Fr, H_, W2_, H_, 192, 1, W2_, 1, W2_, 1,
              0, HW2_, HW2_, B_ * C_, fs, 1, H_);
  launch_gemm(stream, Ch, I1, Fi, H_, W2_, H_, 192, 1, W2_, 1, W2_, 1,
              0, HW2_, HW2_, B_ * C_, fs, 0, H_);
  launch_gemm(stream, Sh, R1, Fi, H_, W2_, H_, 192, 1, W2_, 1, W2_, 1,
              0, HW2_, HW2_, B_ * C_, -fs, 1, H_);
  fftpack_kernel<<<nblk(E_f2), 256, 0, stream>>>(Fr, Fi, bn_g, bn_b, frb);
  dwconv3x3_kernel<<<nblk(E_f2), 256, 0, stream>>>(frb, fpe_w, fpe_b, frb2, 128, H_, W2_, 1);
  launch_gemm(stream, fdc_w, frb2, fdt, 128, (int)HW2_, 128,
              128, 1, (int)HW2_, 1, (int)HW2_, 1, 0, 128 * HW2_, 128 * HW2_, B_, 1.f, 0, 128);
  gelu_deint_kernel<<<nblk(E_f2), 256, 0, stream>>>(fdt, fdc_b, Gr, Gi);
  // inverse FFT along H
  launch_gemm(stream, Ch, Gr, R1i, H_, W2_, H_, 192, 1, W2_, 1, W2_, 1,
              0, HW2_, HW2_, B_ * C_, fs, 0, H_);
  launch_gemm(stream, Sh, Gi, R1i, H_, W2_, H_, 192, 1, W2_, 1, W2_, 1,
              0, HW2_, HW2_, B_ * C_, -fs, 1, H_);
  launch_gemm(stream, Sh, Gr, I1i, H_, W2_, H_, 192, 1, W2_, 1, W2_, 1,
              0, HW2_, HW2_, B_ * C_, fs, 0, H_);
  launch_gemm(stream, Ch, Gi, I1i, H_, W2_, H_, 192, 1, W2_, 1, W2_, 1,
              0, HW2_, HW2_, B_ * C_, fs, 1, H_);
  // irfft along W (Nyquist/DC weights folded into Cw2/Sw2)
  launch_gemm(stream, R1i, Cw2, outf, B_ * C_ * H_, W_, W2_,
              W2_, 1, W_, 1, W_, 1, 0, 0, 0, 1, 1.f, 0, W2_);
  launch_gemm(stream, I1i, Sw2, outf, B_ * C_ * H_, W_, W2_,
              W2_, 1, W_, 1, W_, 1, 0, 0, 0, 1, -1.f, 1, W2_);

  // ---- xc branch + final ----
  launch_gemm(stream, xc_w, x2, q, C_, (int)HW_, C_,
              C_, 1, (int)HW_, 1, (int)HW_, 1, 0, C_ * HW_, C_ * HW_, B_, 1.f, 0, C_);
  final_kernel<<<nblk(E_img), 256, 0, stream>>>(q, xc_b, bn2_g, bn2_b, outf, out);
}
